// TAST_89343909691533
// MI455X (gfx1250) — matrix operand order repaired and checked by reading, not compiled, not run on hardware
//
#include <hip/hip_runtime.h>
#include <hip/hip_bf16.h>

// ---------------------------------------------------------------------------
// Problem constants (from reference setup_inputs)
// ---------------------------------------------------------------------------
constexpr int kB   = 4096;   // batch rows
constexpr int kN   = 3400;   // support rows
constexpr int kNP  = 3456;   // supports padded to multiple of 128
constexpr int kD1  = 2048;   // input feature dim
constexpr int kD2  = 1024;   // ensemble output dim
constexpr int kE   = 10;     // ensemble size
constexpr int kC   = 17;     // classes
constexpr int kTK  = 20;     // top-k
constexpr float kTAU = 10.0f;
constexpr float kEPS = 1e-12f;

typedef __attribute__((ext_vector_type(16))) _Float16 v16h;
typedef __attribute__((ext_vector_type(8)))  _Float16 v8h;
typedef __attribute__((ext_vector_type(8)))  float    v8f;
typedef __attribute__((ext_vector_type(4)))  unsigned int v4u;
typedef __attribute__((ext_vector_type(8)))  int          v8i;
typedef __attribute__((ext_vector_type(4)))  int          v4i;

// Tensor Data Mover availability (device pass only; host pass falls back)
#if defined(__has_builtin)
#if __has_builtin(__builtin_amdgcn_tensor_load_to_lds)
#define HAVE_TDM 1
#endif
#endif
#ifndef HAVE_TDM
#define HAVE_TDM 0
#endif

// ---------------------------------------------------------------------------
// TDM: load a [128 rows x 32 K-elements] f16 tile (row stride K elements in
// global) into LDS with 16B padding every 64B row -> 80B LDS row stride.
// D# built per CDNA5 ISA 8.3/8.4 (group0: count/lds/global/type, group1:
// data_size=2B, pad, dims, strides). Groups 2/3 zero (2D tile).
// ---------------------------------------------------------------------------
#if HAVE_TDM
__device__ __forceinline__ void tdm_load_tile(const _Float16* gptr,
                                              unsigned lds_off, int K, int Nw) {
  unsigned long long ga = (unsigned long long)(size_t)gptr;
  v4u g0 = { 1u,                                   // count=1, is_restore=0
             lds_off,                              // lds_addr (bytes)
             (unsigned)(ga & 0xffffffffu),
             (unsigned)((ga >> 32) & 0x1ffffffu) | (2u << 30) };  // type=2
  v8i g1 = { (int)((1u << 16)        // data_size = 1 (2 bytes)
                 | (1u << 20)        // pad_enable
                 | (3u << 22)        // pad_interval: every 16 DWORDs (64B)
                 | (3u << 25)),      // pad_amount: 4 DWORDs (16B)
             (int)(((unsigned)K & 0xffffu) << 16),            // tensor_dim0 lo
             (int)((((unsigned)K >> 16) & 0xffffu)
                 | (((unsigned)Nw & 0xffffu) << 16)),         // dim0 hi | dim1 lo
             (int)((((unsigned)Nw >> 16) & 0xffffu)
                 | (32u << 16)),                              // dim1 hi | tile_dim0=32
             (int)128u,                                       // tile_dim1=128
             (int)(unsigned)K,                                // tensor_dim0_stride lo
             0, 0 };
  v4i gz = {0, 0, 0, 0};
#if __clang_major__ >= 23
  v8i gz8 = {0, 0, 0, 0, 0, 0, 0, 0};
  __builtin_amdgcn_tensor_load_to_lds(g0, g1, gz, gz, gz8, 0);
#else
  __builtin_amdgcn_tensor_load_to_lds(g0, g1, gz, gz, 0);
#endif
}
#endif

// ---------------------------------------------------------------------------
// 1) Row L2-norm -> inverse norm (1 / max(||row||, eps)), one block per row
// ---------------------------------------------------------------------------
__global__ __launch_bounds__(256) void rownorm_inv_kernel(
    const float* __restrict__ X, int L, float* __restrict__ inv_out) {
  const int r = blockIdx.x;
  const float* row = X + (size_t)r * L;
  float ss = 0.f;
  for (int i = threadIdx.x; i < L; i += 256) { float v = row[i]; ss += v * v; }
  __shared__ float red[256];
  red[threadIdx.x] = ss; __syncthreads();
  for (int s = 128; s > 0; s >>= 1) {
    if (threadIdx.x < s) red[threadIdx.x] += red[threadIdx.x + s];
    __syncthreads();
  }
  if (threadIdx.x == 0) inv_out[r] = 1.0f / fmaxf(sqrtf(red[0]), kEPS);
}

// ---------------------------------------------------------------------------
// 2) f32 -> f16 conversions (plain + row-normalized copies)
// ---------------------------------------------------------------------------
__global__ __launch_bounds__(256) void conv_z_kernel(
    const float* __restrict__ z, const float* __restrict__ invr,
    _Float16* __restrict__ zh, _Float16* __restrict__ znh) {
  const int d = blockIdx.x * 256 + threadIdx.x;
  const int b = blockIdx.y;
  const size_t i = (size_t)b * kD1 + d;
  const float v = z[i];
  zh[i]  = (_Float16)v;
  znh[i] = (_Float16)(v * invr[b]);
}

__global__ __launch_bounds__(256) void conv_s_kernel(
    const float* __restrict__ s, const float* __restrict__ invr,
    _Float16* __restrict__ sh, _Float16* __restrict__ snh) {
  const int d = blockIdx.x * 256 + threadIdx.x;
  const int n = blockIdx.y;                       // 0..kNP-1
  const size_t i = (size_t)n * kD1 + d;
  if (n < kN) {
    const float v = s[i];
    sh[i]  = (_Float16)v;
    snh[i] = (_Float16)(v * invr[n]);
  } else {                                        // zero padding rows
    sh[i]  = (_Float16)0.f;
    snh[i] = (_Float16)0.f;
  }
}

// We[e][o][d] = W[o][d] * alpha[e][d]  (f16), grid (D1/256, D2, E)
__global__ __launch_bounds__(256) void build_we_kernel(
    const float* __restrict__ W, const float* __restrict__ alpha,
    _Float16* __restrict__ We) {
  const int d = blockIdx.x * 256 + threadIdx.x;
  const int o = blockIdx.y, e = blockIdx.z;
  We[(((size_t)e * kD2) + o) * kD1 + d] =
      (_Float16)(W[(size_t)o * kD1 + d] * alpha[(size_t)e * kD1 + d]);
}

// ---------------------------------------------------------------------------
// 3) WMMA GEMM:  C[e, m, col] = sum_d A[m,d] * Bm[e, col, d]
//    Block tile 128(M) x 128(N), 8 waves: wave w owns rows [w*16, w*16+16).
//    B panel (128 cols x 32 K) is shared by all waves, staged in LDS by the
//    Tensor Data Mover (double buffered, s_wait_tensorcnt) with an 80B row
//    stride to spread WMMA B-fragment reads across bank groups.
//    B fragments are software-pipelined: DS loads for fragment j+1 are in
//    flight while WMMA j executes; A global loads hide under the TDM wait.
// ---------------------------------------------------------------------------
template <bool GB>
__global__ __launch_bounds__(256) void gemm_wmma_kernel(
    const _Float16* __restrict__ A, const _Float16* __restrict__ Bm,
    float* __restrict__ Co, int K, int Nw, int ldc,
    size_t b_estride, size_t c_estride,
    const float* __restrict__ gamma, const float* __restrict__ bias) {
  constexpr int ROWB = 80;  // 64B of data + 16B pad per LDS row
  __shared__ __align__(16) unsigned char smem[2][128 * ROWB];

  const int e    = blockIdx.z;
  const int wave = threadIdx.x >> 5;
  const int lane = threadIdx.x & 31;
  const int row_base = blockIdx.x * 128 + wave * 16;
  const int col_base = blockIdx.y * 128;

  const _Float16* Bp = Bm + (size_t)e * b_estride + (size_t)col_base * K;
  float*          Cp = Co + (size_t)e * c_estride;

  const int m = lane & 15;   // A row / B col within 16x16 tile
  const int h = lane >> 4;   // half-wave selector
  const _Float16* Arow = A + (size_t)(row_base + m) * K;

  v8f acc[8];
#pragma unroll
  for (int j = 0; j < 8; ++j)
#pragma unroll
    for (int t = 0; t < 8; ++t) acc[j][t] = 0.f;

#if HAVE_TDM
  const bool issuer = (threadIdx.x < 32);  // wave 0 drives the TDM
  if (issuer) tdm_load_tile(Bp, (unsigned)(size_t)&smem[0][0], K, Nw);
#endif

  for (int ks = 0, it = 0; ks < K; ks += 32, ++it) {
    const int cur = it & 1;

    // A fragment global loads issued early: latency overlaps the TDM wait /
    // barrier below. (ISA 16-bit A layout: lanes 0-15 K[0..8)+K[16..24),
    // lanes 16-31 K[8..16)+K[24..32).)
    v8h alo = *(const v8h*)(Arow + ks + 8 * h);
    v8h ahi = *(const v8h*)(Arow + ks + 16 + 8 * h);

#if HAVE_TDM
    if (issuer) {
      if (ks + 32 < K) {
        tdm_load_tile(Bp + ks + 32, (unsigned)(size_t)&smem[cur ^ 1][0], K, Nw);
        __builtin_amdgcn_s_wait_tensorcnt((short)1);  // current tile landed
      } else {
        __builtin_amdgcn_s_wait_tensorcnt((short)0);
      }
    }
    __syncthreads();          // tile visible to all waves
#else
    __syncthreads();
    // Cooperative copy: 512 x 16B chunks, 2 per thread, coalesced in global.
    for (int c = threadIdx.x; c < 512; c += 256) {
      const int row = c >> 2, part = c & 3;
      *(v8h*)(&smem[cur][row * ROWB + part * 16]) =
          *(const v8h*)(Bp + (size_t)row * K + ks + part * 8);
    }
    __syncthreads();
#endif

    v16h a;
#pragma unroll
    for (int i = 0; i < 8; ++i) { a[i] = alo[i]; a[i + 8] = ahi[i]; }

    // B fragment loader: col = j*16 + m, K = 8*h..8*h+8 and 16+8*h..16+8*h+8 within tile
    const unsigned char* bbase = &smem[cur][0] + m * ROWB + 16 * h;
    auto loadB = [&](int j) -> v16h {
      const unsigned char* bp8 = bbase + j * (16 * ROWB);
      v8h blo = *(const v8h*)(bp8);
      v8h bhi = *(const v8h*)(bp8 + 32);
      v16h b;
#pragma unroll
      for (int i = 0; i < 8; ++i) { b[i] = blo[i]; b[i + 8] = bhi[i]; }
      return b;
    };

    // Software pipeline: fragment j+1 loads issue before WMMA j.
    v16h bcur = loadB(0);
#pragma unroll
    for (int j = 0; j < 7; ++j) {
      v16h bnext = loadB(j + 1);
      acc[j] = __builtin_amdgcn_wmma_f32_16x16x32_f16(
          false, a, false, bcur, (short)0, acc[j], false, false);
      bcur = bnext;
    }
    acc[7] = __builtin_amdgcn_wmma_f32_16x16x32_f16(
        false, a, false, bcur, (short)0, acc[7], false, false);

    __syncthreads();          // all waves done with buf[cur] before reuse
  }

  // Epilogue: C/D layout -> VGPR r holds M = r + 8*h, col = lane&15
  const int row_off = h * 8;
#pragma unroll
  for (int j = 0; j < 8; ++j) {
    const int col = col_base + j * 16 + m;
    float g = 1.f, bb = 0.f;
    if (GB) {
      g  = gamma[(size_t)e * kD2 + col];
      bb = bias [(size_t)e * kD2 + col];
    }
#pragma unroll
    for (int r = 0; r < 8; ++r) {
      const int rr = row_base + row_off + r;
      Cp[(size_t)rr * ldc + col] = GB ? (acc[j][r] * g + bb) : acc[j][r];
    }
  }
}

// ---------------------------------------------------------------------------
// 4) Label helpers: per-class column sums, per-row argmax (labels are one-hot)
// ---------------------------------------------------------------------------
__global__ __launch_bounds__(256) void colsum_kernel(
    const float* __restrict__ labels, float* __restrict__ colsum) {
  const int c = blockIdx.x;  // 17 blocks
  float s = 0.f;
  for (int n = threadIdx.x; n < kN; n += 256) s += labels[(size_t)n * kC + c];
  __shared__ float red[256];
  red[threadIdx.x] = s; __syncthreads();
  for (int t = 128; t > 0; t >>= 1) {
    if (threadIdx.x < t) red[threadIdx.x] += red[threadIdx.x + t];
    __syncthreads();
  }
  if (threadIdx.x == 0) colsum[c] = red[0];
}

__global__ __launch_bounds__(256) void yidx_kernel(
    const float* __restrict__ labels, int* __restrict__ yidx) {
  const int n = blockIdx.x * 256 + threadIdx.x;
  if (n >= kN) return;
  const float* r = labels + (size_t)n * kC;
  int bi = 0; float bv = r[0];
  for (int c = 1; c < kC; ++c) if (r[c] > bv) { bv = r[c]; bi = c; }
  yidx[n] = bi;
}

// centroids[e,c,o] = (sum_{n: y[n]==c} mlp_s[e,n,o]) / (colsum[c]+eps)
__global__ __launch_bounds__(256) void centroid_kernel(
    const float* __restrict__ mlp_s, const int* __restrict__ yidx,
    const float* __restrict__ colsum, float* __restrict__ cent) {
  const int e = blockIdx.y;
  const int o = blockIdx.x * 256 + threadIdx.x;
  __shared__ float acc[256 * kC];
  float* my = &acc[threadIdx.x * kC];
  for (int c = 0; c < kC; ++c) my[c] = 0.f;
  const float* base = mlp_s + (size_t)e * kNP * kD2 + o;
  for (int n = 0; n < kN; ++n) my[yidx[n]] += base[(size_t)n * kD2];
  for (int c = 0; c < kC; ++c)
    cent[(((size_t)e * kC) + c) * kD2 + o] = my[c] / (colsum[c] + kEPS);
}

// ---------------------------------------------------------------------------
// 5) Logits: Out[e,r,c] = TAU * inv_nx[e,r] * inv_nc[e,c] * <X_row, cent_row>
//    One wave per (e, r); 17 shfl-reduced dot products over D2=1024.
// ---------------------------------------------------------------------------
__global__ __launch_bounds__(256) void logits_kernel(
    const float* __restrict__ X, const float* __restrict__ inv_nx,
    const float* __restrict__ cent, const float* __restrict__ inv_nc,
    float* __restrict__ Out, int M, int RowsPad) {
  const int gw   = blockIdx.x * 8 + (threadIdx.x >> 5);
  const int lane = threadIdx.x & 31;
  if (gw >= kE * M) return;
  const int e = gw / M, r = gw % M;
  const float* row = X + ((size_t)e * RowsPad + r) * kD2;
  const float inx = inv_nx[(size_t)e * RowsPad + r];
  float x[32];
#pragma unroll
  for (int i = 0; i < 32; ++i) x[i] = row[lane + 32 * i] * inx;
#pragma unroll 1
  for (int c = 0; c < kC; ++c) {
    const float* cr = cent + (((size_t)e * kC) + c) * kD2;
    float d = 0.f;
#pragma unroll
    for (int i = 0; i < 32; ++i) d += x[i] * cr[lane + 32 * i];
    for (int off = 16; off > 0; off >>= 1) d += __shfl_xor(d, off, 32);
    if (lane == 0)
      Out[((size_t)e * M + r) * kC + c] = kTAU * d * inv_nc[e * kC + c];
  }
}

// ---------------------------------------------------------------------------
// 6) Softmax over C=17, one thread per (e,n)
// ---------------------------------------------------------------------------
__global__ __launch_bounds__(256) void softmax_kernel(
    const float* __restrict__ temp, float* __restrict__ soft, int M) {
  const int i = blockIdx.x * 256 + threadIdx.x;
  if (i >= M) return;
  const float* t = temp + (size_t)i * kC;
  float mx = t[0];
  for (int c = 1; c < kC; ++c) mx = fmaxf(mx, t[c]);
  float v[kC], sum = 0.f;
  for (int c = 0; c < kC; ++c) { v[c] = expf(t[c] - mx); sum += v[c]; }
  const float inv = 1.0f / sum;
  float* o = soft + (size_t)i * kC;
  for (int c = 0; c < kC; ++c) o[c] = v[c] * inv;
}

// ---------------------------------------------------------------------------
// 7) Top-20 per row of G (cosine-sim GEMM output), iterative block argmax
// ---------------------------------------------------------------------------
__global__ __launch_bounds__(256) void topk_kernel(
    const float* __restrict__ G, int* __restrict__ idx_out) {
  const int b = blockIdx.x;
  __shared__ float data[kNP];
  __shared__ float rv[256];
  __shared__ int   ri[256];
  const float* row = G + (size_t)b * kNP;
  for (int i = threadIdx.x; i < kNP; i += 256)
    data[i] = (i < kN) ? row[i] : -3.0e38f;
  __syncthreads();
  for (int k = 0; k < kTK; ++k) {
    float bv = -3.0e38f; int bi = 0;
    for (int i = threadIdx.x; i < kNP; i += 256)
      if (data[i] > bv) { bv = data[i]; bi = i; }
    rv[threadIdx.x] = bv; ri[threadIdx.x] = bi; __syncthreads();
    for (int s = 128; s > 0; s >>= 1) {
      if (threadIdx.x < s && rv[threadIdx.x + s] > rv[threadIdx.x]) {
        rv[threadIdx.x] = rv[threadIdx.x + s];
        ri[threadIdx.x] = ri[threadIdx.x + s];
      }
      __syncthreads();
    }
    if (threadIdx.x == 0) { idx_out[b * kTK + k] = ri[0]; data[ri[0]] = -3.0e38f; }
    __syncthreads();
  }
}

// ---------------------------------------------------------------------------
// 8) outputs[b,c] = mean_e ( sum_k soft[e,idx_k,c] / (sum_c' .. + eps) )
//    One wave per b; lanes 0..16 own classes.
// ---------------------------------------------------------------------------
__global__ __launch_bounds__(32) void outputs_kernel(
    const float* __restrict__ soft, const int* __restrict__ topk,
    float* __restrict__ out) {
  const int b = blockIdx.x, lane = threadIdx.x;
  int idx[kTK];
  for (int k = 0; k < kTK; ++k) idx[k] = topk[b * kTK + k];
  float accm = 0.f;
  for (int e = 0; e < kE; ++e) {
    float s = 0.f;
    if (lane < kC)
      for (int k = 0; k < kTK; ++k)
        s += soft[(((size_t)e * kN) + idx[k]) * kC + lane];
    float tot = s;
    for (int off = 16; off > 0; off >>= 1) tot += __shfl_xor(tot, off, 32);
    accm += s / (tot + kEPS);
  }
  if (lane < kC) out[(size_t)b * kC + lane] = accm * (1.0f / kE);
}

// ---------------------------------------------------------------------------
// Launch
// ---------------------------------------------------------------------------
extern "C" void kernel_launch(void* const* d_in, const int* in_sizes, int n_in,
                              void* d_out, int out_size, void* d_ws, size_t ws_size,
                              hipStream_t stream) {
  const float* z        = (const float*)d_in[0];
  const float* supports = (const float*)d_in[1];
  const float* labels   = (const float*)d_in[2];
  const float* weight   = (const float*)d_in[3];
  const float* alpha    = (const float*)d_in[4];
  const float* gamma    = (const float*)d_in[5];
  const float* ebias    = (const float*)d_in[6];
  float* out = (float*)d_out;                    // [B*C outputs] ++ [E*B*C logits]
  float* logits_out = out + (size_t)kB * kC;

  char* ws = (char*)d_ws;
  size_t off = 0;
  auto carve = [&](size_t bytes) -> void* {
    void* p = ws + off;
    off = (off + bytes + 255) & ~(size_t)255;
    return p;
  };
  _Float16* z_h   = (_Float16*)carve((size_t)kB  * kD1 * 2);
  _Float16* s_h   = (_Float16*)carve((size_t)kNP * kD1 * 2);
  _Float16* zn_h  = (_Float16*)carve((size_t)kB  * kD1 * 2);
  _Float16* sn_h  = (_Float16*)carve((size_t)kNP * kD1 * 2);
  _Float16* We_h  = (_Float16*)carve((size_t)kE * kD2 * kD1 * 2);
  float* mlp_z  = (float*)carve((size_t)kE * kB  * kD2 * 4);
  float* mlp_s  = (float*)carve((size_t)kE * kNP * kD2 * 4);
  float* G      = (float*)carve((size_t)kB * kNP * 4);
  float* inv_rz = (float*)carve((size_t)kB * 4);
  float* inv_rs = (float*)carve((size_t)kN * 4);
  float* inv_nz = (float*)carve((size_t)kE * kB * 4);
  float* inv_ns = (float*)carve((size_t)kE * kNP * 4);
  float* colsum = (float*)carve((size_t)kC * 4);
  int*   yidx   = (int*)  carve((size_t)kN * 4);
  float* cent   = (float*)carve((size_t)kE * kC * kD2 * 4);
  float* inv_nc = (float*)carve((size_t)kE * kC * 4);
  float* temp   = (float*)carve((size_t)kE * kN * kC * 4);
  float* soft   = (float*)carve((size_t)kE * kN * kC * 4);
  int*   topk   = (int*)  carve((size_t)kB * kTK * 4);

  // --- feature row norms + f16 conversions ---
  rownorm_inv_kernel<<<kB, 256, 0, stream>>>(z, kD1, inv_rz);
  rownorm_inv_kernel<<<kN, 256, 0, stream>>>(supports, kD1, inv_rs);
  conv_z_kernel<<<dim3(kD1 / 256, kB), 256, 0, stream>>>(z, inv_rz, z_h, zn_h);
  conv_s_kernel<<<dim3(kD1 / 256, kNP), 256, 0, stream>>>(supports, inv_rs, s_h, sn_h);
  build_we_kernel<<<dim3(kD1 / 256, kD2, kE), 256, 0, stream>>>(weight, alpha, We_h);

  // --- label helpers ---
  colsum_kernel<<<kC, 256, 0, stream>>>(labels, colsum);
  yidx_kernel<<<(kN + 255) / 256, 256, 0, stream>>>(labels, yidx);

  // --- heavy WMMA GEMMs (block tile 128x128, TDM-staged B panel) ---
  // mlp_z[e] = (z * alpha_e) @ W^T * gamma_e + bias_e   -> [E, B, D2]
  gemm_wmma_kernel<true><<<dim3(kB / 128, kD2 / 128, kE), 256, 0, stream>>>(
      z_h, We_h, mlp_z, kD1, kD2, kD2,
      (size_t)kD2 * kD1, (size_t)kB * kD2, gamma, ebias);
  // mlp_s[e] -> [E, NP, D2]
  gemm_wmma_kernel<true><<<dim3(kNP / 128, kD2 / 128, kE), 256, 0, stream>>>(
      s_h, We_h, mlp_s, kD1, kD2, kD2,
      (size_t)kD2 * kD1, (size_t)kNP * kD2, gamma, ebias);
  // G = zn @ sn^T (cosine similarities) -> [B, NP]
  gemm_wmma_kernel<false><<<dim3(kB / 128, kNP / 128, 1), 256, 0, stream>>>(
      zn_h, sn_h, G, kD1, kNP, kNP, 0, 0, nullptr, nullptr);

  // --- embedding row norms ---
  rownorm_inv_kernel<<<kE * kB,  256, 0, stream>>>(mlp_z, kD2, inv_nz);
  rownorm_inv_kernel<<<kE * kNP, 256, 0, stream>>>(mlp_s, kD2, inv_ns);

  // --- centroids + norms ---
  centroid_kernel<<<dim3(kD2 / 256, kE), 256, 0, stream>>>(mlp_s, yidx, colsum, cent);
  rownorm_inv_kernel<<<kE * kC, 256, 0, stream>>>(cent, kD2, inv_nc);

  // --- logits (output #2) and temp (supports-vs-centroids) ---
  logits_kernel<<<(kE * kB + 7) / 8, 256, 0, stream>>>(
      mlp_z, inv_nz, cent, inv_nc, logits_out, kB, kB);
  logits_kernel<<<(kE * kN + 7) / 8, 256, 0, stream>>>(
      mlp_s, inv_ns, cent, inv_nc, temp, kN, kNP);

  // --- softmax, top-k, ensemble-mixed outputs (output #1) ---
  softmax_kernel<<<(kE * kN + 255) / 256, 256, 0, stream>>>(temp, soft, kE * kN);
  topk_kernel<<<kB, 256, 0, stream>>>(G, topk);
  outputs_kernel<<<kB, 32, 0, stream>>>(soft, topk, out);

  (void)in_sizes; (void)n_in; (void)out_size; (void)ws_size;
}
